// Net_12481174962824
// MI455X (gfx1250) — 8-bit matrix operand order repaired and checked by reading, compiled, not run on hardware
//
#include <hip/hip_runtime.h>
#include <stdint.h>

typedef int v8i __attribute__((ext_vector_type(8)));

#define BN_EPS 1e-4f

__global__ void pack_conv_b(const float* __restrict__ w, int8_t* __restrict__ dst) {
  int idx = blockIdx.x * 256 + threadIdx.x;
  if (idx >= 9 * 4 * 1024) return;
  int tile = idx >> 10;
  int t = tile >> 2, g = tile & 3;
  int l = (idx >> 5) & 31, v = (idx >> 2) & 7, c = idx & 3;
  int k  = 16 * (v >> 1) + 8 * (l >> 4) + 4 * (v & 1) + c;
  int oc = g * 16 + (l & 15);
  float wv = w[((oc * 64 + k) * 3 + t / 3) * 3 + (t % 3)];
  dst[idx] = (wv >= 0.f) ? (int8_t)1 : (int8_t)(-1);
}

__global__ void pack_fc1_b(const float* __restrict__ w, int8_t* __restrict__ dst) {
  int idx = blockIdx.x * 256 + threadIdx.x;
  if (idx >= 25 * 32 * 1024) return;
  int tile = idx >> 10;
  int s = tile >> 5, g = tile & 31;
  int l = (idx >> 5) & 31, v = (idx >> 2) & 7, c = idx & 3;
  int kloc = 16 * (v >> 1) + 8 * (l >> 4) + 4 * (v & 1) + c;
  int kglob = s * 64 + kloc;
  int oc = g * 16 + (l & 15);
  float wv = w[(size_t)oc * 1600 + kglob];
  dst[idx] = (wv >= 0.f) ? (int8_t)1 : (int8_t)(-1);
}

__global__ __launch_bounds__(256) void conv1_pool_quant(
    const float* __restrict__ x, const float* __restrict__ w1,
    const float* __restrict__ b1, uint8_t* __restrict__ A2p) {
  __shared__ float xs[30 * 30];
  __shared__ float wsg[64 * 9];
  __shared__ float bs[64];
  const int b = blockIdx.x;
  const int tid = threadIdx.x;
  const float* xb = x + (size_t)b * 784;
  for (int i = tid; i < 900; i += 256) {
    int r = i / 30, c = i % 30;
    float v = 0.f;
    if (r >= 1 && r <= 28 && c >= 1 && c <= 28) v = xb[(r - 1) * 28 + (c - 1)];
    xs[i] = v;
  }
  for (int i = tid; i < 576; i += 256) wsg[i] = (w1[i] >= 0.f) ? 1.f : -1.f;
  if (tid < 64) bs[tid] = b1[tid];
  __syncthreads();

  const int pyp = tid >> 4, pxp = tid & 15;
  uint8_t* dst = A2p + (((size_t)b * 256 + tid) << 6);
  if (pyp == 0 || pyp == 15 || pxp == 0 || pxp == 15) {
    #pragma unroll
    for (int i = 0; i < 4; ++i) ((int4*)dst)[i] = make_int4(0, 0, 0, 0);
    return;
  }
  const int py = pyp - 1, px = pxp - 1;
  float xv[4][4];
  #pragma unroll
  for (int r = 0; r < 4; ++r)
    #pragma unroll
    for (int c = 0; c < 4; ++c) xv[r][c] = xs[(2 * py + r) * 30 + (2 * px + c)];
  for (int oc4 = 0; oc4 < 16; ++oc4) {
    unsigned pack = 0u;
    #pragma unroll
    for (int u = 0; u < 4; ++u) {
      int oc = oc4 * 4 + u;
      const float* wo = wsg + oc * 9;
      float mx = -1e30f;
      #pragma unroll
      for (int dy = 0; dy < 2; ++dy)
        #pragma unroll
        for (int dx = 0; dx < 2; ++dx) {
          float s = bs[oc];
          #pragma unroll
          for (int kh = 0; kh < 3; ++kh)
            #pragma unroll
            for (int kw = 0; kw < 3; ++kw) s += wo[kh * 3 + kw] * xv[dy + kh][dx + kw];
          mx = fmaxf(mx, s);
        }
      mx = fminf(fmaxf(mx, 0.f), 1.f);
      pack |= ((unsigned)__float2int_rn(mx * 3.f)) << (8 * u);
    }
    *(unsigned*)(dst + oc4 * 4) = pack;
  }
}

__device__ __forceinline__ v8i load_a_row(const uint8_t* row, int off) {
  int2 d0 = *(const int2*)(row + off);
  int2 d1 = *(const int2*)(row + off + 16);
  int2 d2 = *(const int2*)(row + off + 32);
  int2 d3 = *(const int2*)(row + off + 48);
  v8i a;
  a[0] = d0.x; a[1] = d0.y; a[2] = d1.x; a[3] = d1.y;
  a[4] = d2.x; a[5] = d2.y; a[6] = d3.x; a[7] = d3.y;
  return a;
}

__global__ __launch_bounds__(128) void conv2_wmma(
    const uint8_t* __restrict__ A2p, uint8_t* __restrict__ A3,
    const int8_t* __restrict__ Wp,
    const float* __restrict__ g1, const float* __restrict__ be1,
    const float* __restrict__ m1, const float* __restrict__ v1) {
  __shared__ __align__(32) int8_t Bl[9 * 4 * 1024];
  const int tid = threadIdx.x;
  for (int i = tid; i < (9 * 4 * 1024) / 16; i += 128)
    ((int4*)Bl)[i] = ((const int4*)Wp)[i];
  __syncthreads();

  const int lane = tid & 31, wv = tid >> 5, half = lane >> 4;
  const int Mbase = blockIdx.x * 64 + wv * 16;
  const int mA = Mbase + (lane & 15);
  const int b = mA / 196, rem = mA % 196;
  const int y = rem / 14, xq = rem % 14;
  const uint8_t* base = A2p + (((size_t)b * 256 + y * 16 + xq) << 6) + (half << 3);
  const int8_t* blane = Bl + lane * 32;

  const v8i vzero = {0, 0, 0, 0, 0, 0, 0, 0};
  v8i acc[4];
  #pragma unroll
  for (int g = 0; g < 4; ++g) acc[g] = vzero;

  #pragma unroll
  for (int t = 0; t < 9; ++t) {
    const int off = ((t / 3) * 16 + (t % 3)) << 6;
    v8i a = load_a_row(base, off);
    #pragma unroll
    for (int g = 0; g < 4; ++g) {
      v8i bm = *((const v8i*)(blane + ((t * 4 + g) << 10)));
      acc[g] = __builtin_amdgcn_wmma_i32_16x16x64_iu8(
           false, a,  true, bm, acc[g], false, false);
    }
  }

  const float inv3 = 1.f / 3.f;
  #pragma unroll
  for (int g = 0; g < 4; ++g) {
    const int oc = g * 16 + (lane & 15);
    const float inv = g1[oc] * rsqrtf(v1[oc] + BN_EPS);
    const float mu = m1[oc], bet = be1[oc];
    #pragma unroll
    for (int r = 0; r < 8; ++r) {
      const int m = Mbase + r + 8 * half;
      float val = (float)acc[g][r] * inv3;
      val = (val - mu) * inv + bet;
      val = fminf(fmaxf(val, 0.f), 1.f);
      A3[((size_t)m << 6) + oc] = (uint8_t)__float2int_rn(val * 3.f);
    }
  }
}

__global__ void pool2_u8(const uint8_t* __restrict__ A3, uint8_t* __restrict__ A3p, int n4) {
  int idx = blockIdx.x * 256 + threadIdx.x;
  if (idx >= n4) return;
  int c4 = idx & 15;
  int t = idx >> 4;
  int x7 = t % 7; t /= 7;
  int y7 = t % 7; t /= 7;
  int b = t;
  unsigned mx = 0u;
  #pragma unroll
  for (int dy = 0; dy < 2; ++dy)
    #pragma unroll
    for (int dx = 0; dx < 2; ++dx) {
      unsigned v = *(const unsigned*)(A3 +
          (((size_t)b * 196 + (2 * y7 + dy) * 14 + (2 * x7 + dx)) << 6) + c4 * 4);
      unsigned r = 0u;
      #pragma unroll
      for (int i = 0; i < 4; ++i) {
        unsigned a = (mx >> (8 * i)) & 255u, bb = (v >> (8 * i)) & 255u;
        r |= (a > bb ? a : bb) << (8 * i);
      }
      mx = r;
    }
  *(unsigned*)(A3p + (((size_t)b * 49 + y7 * 7 + x7) << 6) + c4 * 4) = mx;
}

__global__ __launch_bounds__(128) void conv3_wmma(
    const uint8_t* __restrict__ A3p, uint8_t* __restrict__ A4,
    const int8_t* __restrict__ Wp,
    const float* __restrict__ g2, const float* __restrict__ be2,
    const float* __restrict__ m2, const float* __restrict__ v2) {
  __shared__ __align__(32) int8_t Bl[9 * 4 * 1024];
  const int tid = threadIdx.x;
  for (int i = tid; i < (9 * 4 * 1024) / 16; i += 128)
    ((int4*)Bl)[i] = ((const int4*)Wp)[i];
  __syncthreads();

  const int lane = tid & 31, wv = tid >> 5, half = lane >> 4;
  const int Mbase = blockIdx.x * 64 + wv * 16;
  const int mA = Mbase + (lane & 15);
  const int b = mA / 25, rem = mA % 25;
  const int y = rem / 5, xq = rem % 5;
  const uint8_t* base = A3p + (((size_t)b * 49 + y * 7 + xq) << 6) + (half << 3);
  const int8_t* blane = Bl + lane * 32;

  const v8i vzero = {0, 0, 0, 0, 0, 0, 0, 0};
  v8i acc[4];
  #pragma unroll
  for (int g = 0; g < 4; ++g) acc[g] = vzero;

  #pragma unroll
  for (int t = 0; t < 9; ++t) {
    const int off = ((t / 3) * 7 + (t % 3)) << 6;
    v8i a = load_a_row(base, off);
    #pragma unroll
    for (int g = 0; g < 4; ++g) {
      v8i bm = *((const v8i*)(blane + ((t * 4 + g) << 10)));
      acc[g] = __builtin_amdgcn_wmma_i32_16x16x64_iu8(false, a, true, bm, acc[g], false, false);
    }
  }

  const float inv3 = 1.f / 3.f;
  #pragma unroll
  for (int g = 0; g < 4; ++g) {
    const int oc = g * 16 + (lane & 15);
    const float inv = g2[oc] * rsqrtf(v2[oc] + BN_EPS);
    const float mu = m2[oc], bet = be2[oc];
    #pragma unroll
    for (int r = 0; r < 8; ++r) {
      const int m = Mbase + r + 8 * half;
      float val = (float)acc[g][r] * inv3;
      val = (val - mu) * inv + bet;
      val = fminf(fmaxf(val, 0.f), 1.f);
      A4[((size_t)m << 6) + oc] = (uint8_t)__float2int_rn(val * 3.f);
    }
  }
}

__global__ __launch_bounds__(128) void fc1_wmma(
    const uint8_t* __restrict__ A4, float* __restrict__ H1,
    const int8_t* __restrict__ Wp, const float* __restrict__ fb1) {
  __shared__ __align__(32) int8_t Bl[5 * 4 * 1024];
  const int tid = threadIdx.x;
  const int lane = tid & 31, wv = tid >> 5, half = lane >> 4;
  const int nb = blockIdx.y;
  const int Mbase = blockIdx.x * 64 + wv * 16;
  const int mrow = Mbase + (lane & 15);
  const uint8_t* arow = A4 + (size_t)mrow * 1600 + (half << 3);
  const int8_t* blane = Bl + lane * 32;

  const v8i vzero = {0, 0, 0, 0, 0, 0, 0, 0};
  v8i acc[4];
  #pragma unroll
  for (int g = 0; g < 4; ++g) acc[g] = vzero;

  for (int sc = 0; sc < 25; sc += 5) {
    __syncthreads();
    for (int i = tid; i < 5 * 4 * 64; i += 128) {
      int tile = i >> 6, chunk = i & 63;
      int s = sc + (tile >> 2), gg = tile & 3;
      ((int4*)Bl)[i] = ((const int4*)Wp)[(size_t)(s * 32 + nb * 4 + gg) * 64 + chunk];
    }
    __syncthreads();
    #pragma unroll
    for (int s2 = 0; s2 < 5; ++s2) {
      v8i a = load_a_row(arow + (size_t)sc * 64, s2 * 64);
      #pragma unroll
      for (int g = 0; g < 4; ++g) {
        v8i bm = *((const v8i*)(blane + ((s2 * 4 + g) << 10)));
        acc[g] = __builtin_amdgcn_wmma_i32_16x16x64_iu8(false, a, true, bm, acc[g], false, false);
      }
    }
  }

  const float inv3 = 1.f / 3.f;
  #pragma unroll
  for (int g = 0; g < 4; ++g) {
    const int oc = nb * 64 + g * 16 + (lane & 15);
    const float bb = fb1[oc];
    #pragma unroll
    for (int r = 0; r < 8; ++r) {
      const int m = Mbase + r + 8 * half;
      H1[(size_t)m * 512 + oc] = (float)acc[g][r] * inv3 + bb;
    }
  }
}

__global__ __launch_bounds__(256) void fc2_logsoftmax(
    const float* __restrict__ H1, const float* __restrict__ fw2,
    const float* __restrict__ fb2, float* __restrict__ out) {
  __shared__ float Wl[10 * 512];
  for (int i = threadIdx.x; i < 5120; i += 256) Wl[i] = fw2[i];
  __syncthreads();
  const int lane = threadIdx.x & 31, wv = threadIdx.x >> 5;
  const int row = blockIdx.x * 8 + wv;
  const float* h = H1 + (size_t)row * 512;
  float acc[10];
  #pragma unroll
  for (int j = 0; j < 10; ++j) acc[j] = 0.f;
  for (int k = lane; k < 512; k += 32) {
    float hv = h[k];
    #pragma unroll
    for (int j = 0; j < 10; ++j) acc[j] += hv * Wl[j * 512 + k];
  }
  #pragma unroll
  for (int off = 16; off > 0; off >>= 1)
    #pragma unroll
    for (int j = 0; j < 10; ++j) acc[j] += __shfl_down(acc[j], off, 32);
  if (lane == 0) {
    float mx = -1e30f;
    #pragma unroll
    for (int j = 0; j < 10; ++j) { acc[j] += fb2[j]; mx = fmaxf(mx, acc[j]); }
    float se = 0.f;
    #pragma unroll
    for (int j = 0; j < 10; ++j) se += expf(acc[j] - mx);
    float lse = mx + logf(se);
    #pragma unroll
    for (int j = 0; j < 10; ++j) out[(size_t)row * 10 + j] = acc[j] - lse;
  }
}

extern "C" void kernel_launch(void* const* d_in, const int* in_sizes, int n_in,
                              void* d_out, int out_size, void* d_ws, size_t ws_size,
                              hipStream_t stream) {
  (void)in_sizes; (void)n_in; (void)out_size; (void)ws_size;
  const float* x   = (const float*)d_in[0];
  const float* w1  = (const float*)d_in[1];
  const float* b1  = (const float*)d_in[2];
  const float* w2  = (const float*)d_in[3];
  const float* g1  = (const float*)d_in[4];
  const float* be1 = (const float*)d_in[5];
  const float* m1  = (const float*)d_in[6];
  const float* v1  = (const float*)d_in[7];
  const float* w3  = (const float*)d_in[8];
  const float* g2  = (const float*)d_in[9];
  const float* be2 = (const float*)d_in[10];
  const float* m2  = (const float*)d_in[11];
  const float* v2  = (const float*)d_in[12];
  const float* fw1 = (const float*)d_in[13];
  const float* fb1 = (const float*)d_in[14];
  const float* fw2 = (const float*)d_in[15];
  const float* fb2 = (const float*)d_in[16];
  float* out = (float*)d_out;

  uint8_t* ws = (uint8_t*)d_ws;
  const size_t szA2p = 4096ull * 256 * 64;
  const size_t szA3  = 4096ull * 196 * 64;
  uint8_t* A2p = ws;
  uint8_t* A3  = ws + szA2p;
  uint8_t* A3p = ws;
  uint8_t* A4  = ws + szA2p;
  float*   H1  = (float*)(ws + szA2p + szA3);
  int8_t*  Wp2 = (int8_t*)(ws + szA2p + szA3 + 4096ull * 512 * 4);
  int8_t*  Wp3 = Wp2 + 36864;
  int8_t*  Wpf = Wp3 + 36864;

  pack_conv_b<<<144, 256, 0, stream>>>(w2, Wp2);
  pack_conv_b<<<144, 256, 0, stream>>>(w3, Wp3);
  pack_fc1_b<<<3200, 256, 0, stream>>>(fw1, Wpf);

  conv1_pool_quant<<<4096, 256, 0, stream>>>(x, w1, b1, A2p);
  conv2_wmma<<<12544, 128, 0, stream>>>(A2p, A3, Wp2, g1, be1, m1, v1);
  pool2_u8<<<12544, 256, 0, stream>>>(A3, A3p, 4096 * 49 * 16);
  conv3_wmma<<<1600, 128, 0, stream>>>(A3p, A4, Wp3, g2, be2, m2, v2);
  fc1_wmma<<<dim3(64, 8), 128, 0, stream>>>(A4, H1, Wpf, fb1);
  fc2_logsoftmax<<<512, 256, 0, stream>>>(H1, fw2, fb2, out);
}
